// Head_49976239456456
// MI455X (gfx1250) — matrix operand order repaired and checked by reading, not compiled, not run on hardware
//
#include <hip/hip_runtime.h>
#include <hip/hip_bf16.h>

#define BB 4
#define TT 4096
#define DM 256
#define HS 64

typedef __attribute__((ext_vector_type(16))) __bf16 v16bf;
typedef __attribute__((ext_vector_type(8)))  __bf16 v8bf;
typedef __attribute__((ext_vector_type(8)))  float  v8f;
typedef __attribute__((ext_vector_type(4)))  float  v4f;
typedef __attribute__((ext_vector_type(4)))  unsigned int v4u;
typedef __attribute__((ext_vector_type(8)))  int    v8i;
typedef __attribute__((ext_vector_type(4)))  int    v4i;

static __device__ __forceinline__ v8f wmma_bf16(v16bf a, v16bf b, v8f c) {
  // D = A(16x32 bf16) x B(32x16 bf16) + C(16x16 f32)
  return __builtin_amdgcn_wmma_f32_16x16x32_bf16(false, a, false, b, (short)0, c,
                                                 false, false);
}

// TDM: DMA a 32-row x 64-col bf16 tile (row-major, row stride 64) from global
// into LDS. D# per CDNA5 ISA 8.3/8.4: group0 = {flags, lds_addr, gaddr}, group1
// holds dims/strides. 2-D tile -> groups 2/3 zero. Tracked by TENSORcnt.
static __device__ __forceinline__ void tdm_load_tile_32x64(const __bf16* gsrc,
                                                           const __bf16* ldst) {
  unsigned long long ga = (unsigned long long)(uintptr_t)gsrc;
  unsigned int lds_off  = (unsigned int)(uintptr_t)ldst;  // flat LDS addr[31:0] = LDS offset
  v4u g0;
  g0[0] = 1u;                                   // count=1 (valid user descriptor)
  g0[1] = lds_off;                              // lds_addr (bytes)
  g0[2] = (unsigned int)ga;                     // global_addr[31:0]
  g0[3] = ((unsigned int)(ga >> 32) & 0x01FFFFFFu) | (2u << 30);  // addr[56:32] | type=2
  v8i g1;
  g1[0] = (int)(1u << 16);                      // workgroup_mask=0, data_size=1 (2 bytes)
  g1[1] = (int)((unsigned int)HS << 16);        // tensor_dim0 = 64 (bits 79:48, low half)
  g1[2] = (int)((unsigned int)TT << 16);        // tensor_dim0 hi=0 | tensor_dim1 = 4096 (lo16)
  g1[3] = (int)((unsigned int)HS << 16);        // tensor_dim1 hi=0 | tile_dim0 = 64
  g1[4] = 32;                                   // tile_dim1 = 32 rows, tile_dim2 = 0
  g1[5] = HS;                                   // tensor_dim0_stride = 64 elements
  g1[6] = 0;
  g1[7] = 0;
  v4i z4 = {};
  v8i z8 = {};
  __builtin_amdgcn_tensor_load_to_lds(g0, g1, z4, z4, z8, 0);
}

// ---------------------------------------------------------------------------
// Kernel 1: fused K/Q/V projections. One wave = one (batch, 16-row t-tile);
// x rows are loaded ONCE into bf16 A tiles (K=256 in 8 chunks), each W is
// staged fp32->bf16 into LDS once per block and consumed via ds_load_b128.
// 96 WMMAs per wave. Output bf16 [3][B,T,64] in d_ws.
// ---------------------------------------------------------------------------
__global__ __launch_bounds__(256) void proj_kernel(
    const float* __restrict__ x,
    const float* __restrict__ Wk, const float* __restrict__ bk,
    const float* __restrict__ Wq, const float* __restrict__ bq,
    const float* __restrict__ Wv, const float* __restrict__ bv,
    __bf16* __restrict__ ws) {
  __shared__ __align__(16) __bf16 Wl[DM * HS];  // 32 KB

  const int lane = threadIdx.x & 31;
  const int wv   = threadIdx.x >> 5;
  const int wid  = blockIdx.x * 8 + wv;         // 1024 tile-waves
  const int b    = wid / (TT / 16);
  const int t0   = (wid % (TT / 16)) * 16;
  const int m    = lane & 15;
  const int hi   = lane >> 4;
  const size_t projStride = (size_t)BB * TT * HS;

  // Preload this wave's x rows as A tiles: M = t = lane%16, K = c (split halves)
  v16bf xa[8];
  const float* xrow = x + ((size_t)b * TT + (t0 + m)) * DM;
#pragma unroll
  for (int k = 0; k < 8; ++k) {
    const int c0 = k * 32 + 8 * hi;
    const v4f* xp0 = (const v4f*)(xrow + c0);
    const v4f* xp1 = (const v4f*)(xrow + c0 + 16);
    v4f f0 = xp0[0], f1 = xp0[1], f2 = xp1[0], f3 = xp1[1];
#pragma unroll
    for (int e = 0; e < 4; ++e) {
      xa[k][e]      = (__bf16)f0[e];
      xa[k][4 + e]  = (__bf16)f1[e];
      xa[k][8 + e]  = (__bf16)f2[e];
      xa[k][12 + e] = (__bf16)f3[e];
    }
  }

#pragma unroll
  for (int p = 0; p < 3; ++p) {
    const float* W    = (p == 0) ? Wk : (p == 1) ? Wq : Wv;
    const float* bias = (p == 0) ? bk : (p == 1) ? bq : bv;
    __bf16* dst       = ws + (size_t)p * projStride;

    __syncthreads();  // previous projection's Wl reads complete
    for (int i = threadIdx.x; i < DM * HS / 4; i += 256) {
      v4f f = ((const v4f*)W)[i];
#pragma unroll
      for (int e = 0; e < 4; ++e) Wl[((4 * i + e) % HS) * DM + (4 * i + e) / HS] = (__bf16)f[e];
    }
    __syncthreads();

    v8f acc[4] = {};
#pragma unroll
    for (int k = 0; k < 8; ++k) {
      const __bf16* wcol = Wl + k * 32 + 8 * hi;  // Wl is [h][c]; lane%16 = N, elements = K (split halves)
#pragma unroll
      for (int j = 0; j < 4; ++j) {
        v8bf lo = *(const v8bf*)(wcol + (size_t)(16 * j + m) * DM);
        v8bf hh = *(const v8bf*)(wcol + (size_t)(16 * j + m) * DM + 16);
        v16bf bt;
#pragma unroll
        for (int e = 0; e < 8; ++e) { bt[e] = lo[e]; bt[8 + e] = hh[e]; }
        acc[j] = wmma_bf16(xa[k], bt, acc[j]);
      }
    }

    // D layout: VGPR r -> t = t0 + r + 8*hi ; N = h = 16*j + m
#pragma unroll
    for (int j = 0; j < 4; ++j) {
      float bj = bias[16 * j + m];
#pragma unroll
      for (int r = 0; r < 8; ++r) {
        int t = t0 + r + 8 * hi;
        dst[((size_t)b * TT + t) * HS + 16 * j + m] = (__bf16)(acc[j][r] + bj);
      }
    }
  }
}

// ---------------------------------------------------------------------------
// Kernel 2: flash attention. 8 waves/block share one batch; each wave owns a
// 16-row t-tile (block covers 128 t rows). Q and V s-tiles (32x64 bf16) are
// DMA'd into LDS by the Tensor Data Mover (double-buffered, overlapped with
// compute), so each tile is read from L2 once per block instead of 8x.
// Scores computed transposed (D[s][t]) so softmax is register+shfl_xor(16)
// and the P tile is directly the A operand of the P*V WMMA.
// ---------------------------------------------------------------------------
__global__ __launch_bounds__(256) void attn_kernel(const __bf16* __restrict__ ws,
                                                   float* __restrict__ out) {
  __shared__ __align__(16) __bf16 ldsQ[2][32 * HS];  // 2 x 4 KB
  __shared__ __align__(16) __bf16 ldsV[2][32 * HS];  // 2 x 4 KB

  const size_t projStride = (size_t)BB * TT * HS;
  const __bf16* Kb = ws;
  const __bf16* Qb = ws + projStride;
  const __bf16* Vb = ws + 2 * projStride;

  const int lane = threadIdx.x & 31;
  const int wv   = threadIdx.x >> 5;
  const int b    = blockIdx.x >> 5;                    // 32 blocks per batch
  const int t0   = ((blockIdx.x & 31) << 7) + wv * 16; // 128 t rows per block
  const int m    = lane & 15;
  const int hi   = lane >> 4;

  const __bf16* qb = Qb + (size_t)b * TT * HS;
  const __bf16* vb = Vb + (size_t)b * TT * HS;

  // K^T as B operand: lane%16 = n = t0+m, element e = k-dim h (split halves)
  v16bf kb0, kb1;
  {
    const __bf16* kbase = Kb + ((size_t)b * TT + t0) * HS;
#pragma unroll
    for (int e = 0; e < 16; ++e) {
      kb0[e] = kbase[(size_t)m * HS + 8 * hi + (e & 7) + 16 * (e >> 3)];
      kb1[e] = kbase[(size_t)m * HS + 32 + 8 * hi + (e & 7) + 16 * (e >> 3)];
    }
  }

  v8f acc[4] = {};
  float m_run = -__builtin_inff();
  float l_run = 0.0f;
  const float scale = 0.0625f;  // 256^-0.5

  if (wv == 0) {  // prologue: DMA first s-block into buffer 0
    tdm_load_tile_32x64(qb, &ldsQ[0][0]);
    tdm_load_tile_32x64(vb, &ldsV[0][0]);
  }

  for (int it = 0; it < TT / 32; ++it) {
    const int s0  = it * 32;
    const int cur = it & 1;
    if (wv == 0) __builtin_amdgcn_s_wait_tensorcnt(0);  // current buffer ready
    __syncthreads();
    if (wv == 0 && s0 + 32 < TT) {  // prefetch next s-block into other buffer
      tdm_load_tile_32x64(qb + (size_t)(s0 + 32) * HS, &ldsQ[cur ^ 1][0]);
      tdm_load_tile_32x64(vb + (size_t)(s0 + 32) * HS, &ldsV[cur ^ 1][0]);
    }
    const __bf16* Qs = &ldsQ[cur][0];
    const __bf16* Vs = &ldsV[cur][0];

    // Q as A operand from LDS: M = s row, K = h (split halves)
    v16bf qa[2][2];
#pragma unroll
    for (int ss = 0; ss < 2; ++ss) {
      const __bf16* qrow = Qs + (size_t)(16 * ss + m) * HS;
#pragma unroll
      for (int hc = 0; hc < 2; ++hc) {
        const int c0 = 32 * hc + 8 * hi;
        v8bf lo = *(const v8bf*)(qrow + c0);
        v8bf hh = *(const v8bf*)(qrow + c0 + 16);
#pragma unroll
        for (int e = 0; e < 8; ++e) { qa[ss][hc][e] = lo[e]; qa[ss][hc][8 + e] = hh[e]; }
      }
    }
    v8f z = {};
    v8f st0 = wmma_bf16(qa[0][0], kb0, z);  // s in [s0, s0+16)
    st0     = wmma_bf16(qa[0][1], kb1, st0);
    v8f st1 = wmma_bf16(qa[1][0], kb0, z);  // s in [s0+16, s0+32)
    st1     = wmma_bf16(qa[1][1], kb1, st1);

    // online softmax over s for column t = lane%16
    float sv0[8], sv1[8];
    float bm = -__builtin_inff();
#pragma unroll
    for (int r = 0; r < 8; ++r) {
      sv0[r] = st0[r] * scale;
      sv1[r] = st1[r] * scale;
      bm = fmaxf(bm, fmaxf(sv0[r], sv1[r]));
    }
    bm = fmaxf(bm, __shfl_xor(bm, 16, 32));
    float m_new = fmaxf(m_run, bm);
    float corr  = __expf(m_run - m_new);  // 0 on first block
    m_run = m_new;

    // P tile -> A operand directly (reg r of st0/st1 == A element e / e+8)
    v16bf pa;
    float lsum = 0.0f;
#pragma unroll
    for (int r = 0; r < 8; ++r) {
      float p0 = __expf(sv0[r] - m_new);
      float p1 = __expf(sv1[r] - m_new);
      lsum += p0 + p1;
      pa[r]     = (__bf16)p0;
      pa[8 + r] = (__bf16)p1;
    }
    lsum += __shfl_xor(lsum, 16, 32);
    l_run = l_run * corr + lsum;

    // rescale accumulators: acc row t = r + 8*hi; its corr lives in lane t
#pragma unroll
    for (int r = 0; r < 8; ++r) {
      float cr = __shfl(corr, 8 * hi + r, 32);
#pragma unroll
      for (int j = 0; j < 4; ++j) acc[j][r] *= cr;
    }

    // V as B operand from LDS: lane%16 = n = h, element e = k-dim s (split halves)
#pragma unroll
    for (int j = 0; j < 4; ++j) {
      v16bf vt;
#pragma unroll
      for (int e = 0; e < 16; ++e)
        vt[e] = Vs[(size_t)(8 * hi + (e & 7) + 16 * (e >> 3)) * HS + 16 * j + m];
      acc[j] = wmma_bf16(pa, vt, acc[j]);
    }
    __syncthreads();  // all waves done with this buffer before it is re-filled
  }

  // epilogue: divide by row sum; l for row t is in lane t
  float* orow = out + ((size_t)b * TT + t0) * HS;
#pragma unroll
  for (int r = 0; r < 8; ++r) {
    float li = 1.0f / __shfl(l_run, 8 * hi + r, 32);
    int t = r + 8 * hi;
#pragma unroll
    for (int j = 0; j < 4; ++j)
      orow[(size_t)t * HS + 16 * j + m] = acc[j][r] * li;
  }
}

extern "C" void kernel_launch(void* const* d_in, const int* in_sizes, int n_in,
                              void* d_out, int out_size, void* d_ws, size_t ws_size,
                              hipStream_t stream) {
  (void)in_sizes; (void)n_in; (void)out_size; (void)ws_size;
  const float* x  = (const float*)d_in[0];
  const float* Wk = (const float*)d_in[1];
  const float* bk = (const float*)d_in[2];
  const float* Wq = (const float*)d_in[3];
  const float* bq = (const float*)d_in[4];
  const float* Wv = (const float*)d_in[5];
  const float* bv = (const float*)d_in[6];
  float* out = (float*)d_out;
  __bf16* ws = (__bf16*)d_ws;  // 3 * B*T*H bf16 = 6 MB

  // fused projections: 4 * 256 = 1024 tile-waves, 8 waves/block
  proj_kernel<<<1024 / 8, 256, 0, stream>>>(x, Wk, bk, Wq, bq, Wv, bv, ws);
  // attention: 128 blocks, each = 8 t-tiles of one batch sharing TDM-staged LDS
  attn_kernel<<<BB * (TT / 128), 256, 0, stream>>>(ws, out);
}
